// AugmentedConv_120259084815
// MI455X (gfx1250) — compile-verified, matrix operand order checked by reading, not run on hardware
//
#include <hip/hip_runtime.h>
#include <hip/hip_bf16.h>

typedef __attribute__((ext_vector_type(16))) _Float16 v16h;
typedef __attribute__((ext_vector_type(8)))  float    v8f;
typedef __attribute__((ext_vector_type(2)))  float    v2f;
typedef __attribute__((address_space(3)))    _Float16 lds_f16;

// Fragment K-offset for 16-bit WMMA A/B operands (ISA 7.12.2, 16-bit 16x32 layout):
// lane = hf*16 + (m|n); element e -> K = (e<8) ? hf*8+e : 16+hf*8+(e-8)
__device__ __forceinline__ int kof(int e, int hf) {
    return (e < 8) ? (hf * 8 + e) : (16 + hf * 8 + (e - 8));
}

// ---------------------------------------------------------------------------
// Kernel 1a: wfrag = combined conv weights (conv_out 64ch ++ qkv 192ch),
// pre-swizzled into exact B-fragment order:
//   wfrag[((kt*256 + col)*2 + hf)*16 + e] = W[K = kt*32 + kof(e,hf)][col]
// ---------------------------------------------------------------------------
__global__ void prep_w_kernel(const float* __restrict__ wco,
                              const float* __restrict__ wqkv,
                              _Float16* __restrict__ wfrag) {
    int j = blockIdx.x * blockDim.x + threadIdx.x;
    if (j >= 147456) return;                       // 18*256*2*16
    int e   = j & 15;
    int hf  = (j >> 4) & 1;
    int col = (j >> 5) & 255;
    int kt  = j >> 13;
    int K = kt * 32 + kof(e, hf);                  // < 576 always
    int c = K / 9, r = K - c * 9;
    float v = (col < 64) ? wco[(col * 64 + c) * 9 + r]
                         : wqkv[((col - 64) * 64 + c) * 9 + r];
    wfrag[j] = (_Float16)v;
}

// ---------------------------------------------------------------------------
// Kernel 1b: afrag = im2col(x) in exact A-fragment order, f16:
//   afrag[((mt*18 + kt)*32 + lane)*16 + e] = im2col[p = mt*16 + lane%16]
//                                                  [K = kt*32 + kof(e, lane/16)]
// ---------------------------------------------------------------------------
__global__ void prep_a_kernel(const float* __restrict__ x,
                              _Float16* __restrict__ afrag) {
    int i = blockIdx.x * blockDim.x + threadIdx.x;
    if (i >= 4718592) return;                      // 512*18*32*16
    int e    = i & 15;
    int lane = (i >> 4) & 31;
    int t    = i >> 9;
    int mt   = t / 18;
    int kt   = t - mt * 18;
    int hf = lane >> 4, m = lane & 15;
    int p = mt * 16 + m;
    int b = p >> 10, yy = (p >> 5) & 31, xx = p & 31;
    int K = kt * 32 + kof(e, hf);
    int c = K / 9, r = K - c * 9;
    int ky = r / 3, kx = r - ky * 3;
    int iy = yy + ky - 1, ix = xx + kx - 1;
    float v = 0.f;
    if (iy >= 0 && iy < 32 && ix >= 0 && ix < 32)
        v = x[((b * 64 + c) * 32 + iy) * 32 + ix];
    afrag[i] = (_Float16)v;
}

// ---------------------------------------------------------------------------
// Kernel 2: fused implicit-GEMM 3x3 convs. M=8192 pixels, N=256 ch, K=576.
// Each wave: 16(M) x 64(N) tile, f16 WMMA, 18 K-steps of 32.
// Per K-step the 16KB B tile (shared by all 8 waves of the block) is staged
// into LDS via global_load_async_to_lds_b128 (ASYNCcnt path). The LDS dest
// is passed to the asm as an addrspace(3) POINTER (32-bit VGPR = LDS byte
// address) so the buffer escapes and the "memory" clobber correctly orders
// the ds_load_b128 fragment reads against the async writes.
// ---------------------------------------------------------------------------
__global__ void conv_gemm_kernel(const _Float16* __restrict__ afrag,
                                 const _Float16* __restrict__ wfrag,
                                 const float* __restrict__ bco,
                                 const float* __restrict__ bqkv,
                                 float* __restrict__ out,
                                 float* __restrict__ q_ws,
                                 float* __restrict__ k_ws,
                                 _Float16* __restrict__ v_ws) {
    __shared__ _Float16 lb[8192];                  // 16KB: B tile of current kt
    int tid  = blockIdx.x * blockDim.x + threadIdx.x;
    int wid  = tid >> 5;
    int lane = threadIdx.x & 31;
    int mt   = wid >> 2;          // 512 M-tiles (2 per block)
    int nt4  = wid & 3;           // 4 N-groups of 64 (also: output category)
    int hf   = lane >> 4, mn = lane & 15;

    v8f c0 = {}, c1 = {}, c2 = {}, c3 = {};
    for (int kt = 0; kt < 18; ++kt) {
        // ---- async-stage B tile: 256 threads x 4 x 16B = 16KB ----
        const _Float16* src = wfrag + (size_t)kt * 8192;
#pragma unroll
        for (int r = 0; r < 4; ++r) {
            lds_f16* lp = (lds_f16*)(lb + (size_t)(threadIdx.x + r * 256) * 8);
            const _Float16* g = src + (size_t)(threadIdx.x + r * 256) * 8;
            asm volatile("global_load_async_to_lds_b128 %0, %1, off"
                         :: "v"(lp), "v"(g) : "memory");
        }
        asm volatile("s_wait_asynccnt 0x0" ::: "memory");
        __syncthreads();

        // ---- A fragment: one aligned 32B load ----
        const _Float16* ap = afrag + ((size_t)(mt * 18 + kt) * 32 + lane) * 16;
        v16h a = *(const v16h*)ap;
        if (kt + 1 < 18)
            __builtin_prefetch(ap + 512, 0, 3);    // next kt A fragment

        // ---- 4 B fragments, then 4 WMMAs ----
        const _Float16* lbase = lb + (size_t)(nt4 * 64 + mn) * 32 + hf * 16;
        v16h b0 = *(const v16h*)(lbase);
        v16h b1 = *(const v16h*)(lbase + 512);     // +16 cols * 32
        v16h b2 = *(const v16h*)(lbase + 1024);
        v16h b3 = *(const v16h*)(lbase + 1536);
        c0 = __builtin_amdgcn_wmma_f32_16x16x32_f16(false, a, false, b0, (short)0, c0, false, false);
        c1 = __builtin_amdgcn_wmma_f32_16x16x32_f16(false, a, false, b1, (short)0, c1, false, false);
        c2 = __builtin_amdgcn_wmma_f32_16x16x32_f16(false, a, false, b2, (short)0, c2, false, false);
        c3 = __builtin_amdgcn_wmma_f32_16x16x32_f16(false, a, false, b3, (short)0, c3, false, false);
        __syncthreads();                           // protect lb before next stage
    }

    // ---- wave-uniform writeback ----
    const float QSCALE = 0.35355339059327373f;     // 8^-0.5
    v8f acc[4] = { c0, c1, c2, c3 };
    if (nt4 == 0) {                                // conv_out + bias -> d_out ch 0..63
#pragma unroll
        for (int t = 0; t < 4; ++t) {
            int col = t * 16 + mn;
#pragma unroll
            for (int j = 0; j < 8; ++j) {
                int pp = mt * 16 + j + 8 * hf;
                int bb = pp >> 10, prow = pp & 1023;
                out[(size_t)(bb * 128 + col) * 1024 + prow] = acc[t][j] + bco[col];
            }
        }
    } else if (nt4 == 1) {                         // q (scaled) -> q_ws[b,h,p,d]
#pragma unroll
        for (int t = 0; t < 4; ++t) {
            int ch = t * 16 + mn, h = ch >> 3, d = ch & 7;
#pragma unroll
            for (int j = 0; j < 8; ++j) {
                int pp = mt * 16 + j + 8 * hf;
                int bb = pp >> 10, prow = pp & 1023;
                q_ws[(size_t)((bb * 8 + h) * 1024 + prow) * 8 + d] =
                    (acc[t][j] + bqkv[ch]) * QSCALE;
            }
        }
    } else if (nt4 == 2) {                         // k -> k_ws[b,h,d,p] (f32)
#pragma unroll
        for (int t = 0; t < 4; ++t) {
            int ch = t * 16 + mn, h = ch >> 3, d = ch & 7;
#pragma unroll
            for (int j = 0; j < 8; ++j) {
                int pp = mt * 16 + j + 8 * hf;
                int bb = pp >> 10, prow = pp & 1023;
                k_ws[(size_t)((bb * 8 + h) * 8 + d) * 1024 + prow] =
                    acc[t][j] + bqkv[64 + ch];
            }
        }
    } else {                                       // v -> v_ws[b,h,d,p] (f16)
#pragma unroll
        for (int t = 0; t < 4; ++t) {
            int ch = t * 16 + mn, h = ch >> 3, d = ch & 7;
#pragma unroll
            for (int j = 0; j < 8; ++j) {
                int pp = mt * 16 + j + 8 * hf;
                int bb = pp >> 10, prow = pp & 1023;
                v_ws[(size_t)((bb * 8 + h) * 8 + d) * 1024 + prow] =
                    (_Float16)(acc[t][j] + bqkv[128 + ch]);
            }
        }
    }
}

// ---------------------------------------------------------------------------
// Kernel 3: relative logits, analytic skew form.
//   RW[b,h,qy,qx,kx] = sum_d q[b,h,qy*32+qx,d] * key_rel_w[kx-qx+31, d]
//   RH[b,h,qx,qy,ky] = sum_d q[b,h,qy*32+qx,d] * key_rel_h[ky-qy+31, d]
// ---------------------------------------------------------------------------
__global__ void rel_kernel(const float* __restrict__ q_ws,
                           const float* __restrict__ krw,
                           const float* __restrict__ krh,
                           float* __restrict__ RW, float* __restrict__ RH) {
    int i = blockIdx.x * blockDim.x + threadIdx.x;
    bool isW = i < 2097152;
    int idx = isW ? i : i - 2097152;
    if (idx >= 2097152) return;
    int k_ = idx & 31;          // kx (W) / ky (H)
    int a2 = (idx >> 5) & 31;   // qx (W) / qy (H)
    int a1 = (idx >> 10) & 31;  // qy (W) / qx (H)
    int h = (idx >> 15) & 7, b = idx >> 18;
    int p = isW ? (a1 * 32 + a2) : (a2 * 32 + a1);
    int m = k_ - a2 + 31;       // always in [0,62]
    const float* qrow = q_ws + (size_t)((b * 8 + h) * 1024 + p) * 8;
    const float* rk   = (isW ? krw : krh) + m * 8;
    float s = 0.f;
#pragma unroll
    for (int d = 0; d < 8; ++d) s += qrow[d] * rk[d];
    if (isW) RW[idx] = s; else RH[idx] = s;
}

// ---------------------------------------------------------------------------
// Kernel 4: attention. One WG per (b, h, 32-row q block).
// Dynamic LDS: S = 32x1024 f32 (128KB) + P = 32x1024 f16 (64KB) = 192KB.
//  Phase 1: S = QK^T (f32 WMMA 16x16x4, K=8 in 2 steps) + RW + RH -> LDS
//  Phase 2: softmax, 2 threads/row, float4 reads, shfl_xor combines -> P (f16)
//  Phase 3: O = P.V  (f16 WMMA 16x16x32, K=1024 in 32 steps, N padded 8->16)
// ---------------------------------------------------------------------------
__global__ void attn_kernel(const float* __restrict__ q_ws,
                            const float* __restrict__ k_ws,
                            const _Float16* __restrict__ v_ws,
                            const float* __restrict__ RW,
                            const float* __restrict__ RH,
                            float* __restrict__ attn_pre) {
    extern __shared__ char smem[];
    float*    S  = (float*)smem;                    // 32*1024 f32
    _Float16* Ph = (_Float16*)(smem + 131072);      // 32*1024 f16

    int bh = blockIdx.x >> 5;     // 64 (b,h) pairs
    int qb = blockIdx.x & 31;     // 32 q-blocks of 32 rows
    int b = bh >> 3, h = bh & 7;
    int wave = threadIdx.x >> 5, lane = threadIdx.x & 31;
    int hf = lane >> 4, mn = lane & 15;
    int qr = qb * 32 + wave * 16;           // image-local q row base of this wave

    const float* qbase = q_ws + (size_t)(b * 8 + h) * 1024 * 8;
    const float* kbase = k_ws + (size_t)(b * 8 + h) * 8 * 1024;

    // -------- Phase 1: scores --------
    float a00 = qbase[(size_t)(qr + mn) * 8 + hf * 2];
    float a01 = qbase[(size_t)(qr + mn) * 8 + hf * 2 + 1];
    float a10 = qbase[(size_t)(qr + mn) * 8 + 4 + hf * 2];
    float a11 = qbase[(size_t)(qr + mn) * 8 + 4 + hf * 2 + 1];

    for (int kt = 0; kt < 64; ++kt) {
        v8f c = {};
        {
            v2f a, bb;
            int d0 = hf * 2;
            a.x  = a00; a.y = a01;
            bb.x = kbase[(size_t)d0 * 1024 + kt * 16 + mn];
            bb.y = kbase[(size_t)(d0 + 1) * 1024 + kt * 16 + mn];
            c = __builtin_amdgcn_wmma_f32_16x16x4_f32(
                false, a, false, bb, (short)0, c, false, false);
            int d1 = 4 + hf * 2;
            a.x  = a10; a.y = a11;
            bb.x = kbase[(size_t)d1 * 1024 + kt * 16 + mn];
            bb.y = kbase[(size_t)(d1 + 1) * 1024 + kt * 16 + mn];
            c = __builtin_amdgcn_wmma_f32_16x16x4_f32(
                false, a, false, bb, (short)0, c, false, false);
        }
        int kidx = kt * 16 + mn, ky = kidx >> 5, kx = kidx & 31;
#pragma unroll
        for (int j = 0; j < 8; ++j) {
            int m = j + 8 * hf;
            int p = qr + m, qy = p >> 5, qx = p & 31;
            float rel = RW[(size_t)(((b * 8 + h) * 32 + qy) * 32 + qx) * 32 + kx] +
                        RH[(size_t)(((b * 8 + h) * 32 + qx) * 32 + qy) * 32 + ky];
            S[(size_t)(wave * 16 + m) * 1024 + kidx] = c[j] + rel;
        }
    }
    __syncthreads();

    // -------- Phase 2: softmax, 2 threads per row --------
    {
        int r = threadIdx.x >> 1, hs = threadIdx.x & 1;
        float* row = S + (size_t)r * 1024;
        float4* rv = (float4*)(row + hs * 512);     // 128 float4 per half-row
        float mx = -3.4e38f;
        for (int i = 0; i < 128; ++i) {
            float4 v = rv[i];
            mx = fmaxf(mx, fmaxf(fmaxf(v.x, v.y), fmaxf(v.z, v.w)));
        }
        mx = fmaxf(mx, __shfl_xor(mx, 1));
        float sum = 0.f;
        for (int i = 0; i < 128; ++i) {
            float4 v = rv[i];
            v.x = __expf(v.x - mx); v.y = __expf(v.y - mx);
            v.z = __expf(v.z - mx); v.w = __expf(v.w - mx);
            rv[i] = v;
            sum += (v.x + v.y) + (v.z + v.w);
        }
        sum += __shfl_xor(sum, 1);
        float inv = 1.f / sum;
        const float*  src  = row + hs * 512;
        _Float16*     prow = Ph + (size_t)r * 1024 + hs * 512;
        for (int i = 0; i < 512; ++i) prow[i] = (_Float16)(src[i] * inv);
    }
    __syncthreads();

    // -------- Phase 3: O = P.V --------
    {
        v8f c = {};
        const _Float16* vbase = v_ws + (size_t)(b * 8 + h) * 8 * 1024;
        const _Float16* arow  = Ph + (size_t)(wave * 16 + mn) * 1024;
        for (int kt = 0; kt < 32; ++kt) {
            int k0 = kt * 32;
            union { uint4 u[2]; v16h v; } ua, ub;
            ua.u[0] = *(const uint4*)(arow + k0 + hf * 8);
            ua.u[1] = *(const uint4*)(arow + k0 + 16 + hf * 8);
            if (mn < 8) {
                const _Float16* vrow = vbase + (size_t)mn * 1024;
                ub.u[0] = *(const uint4*)(vrow + k0 + hf * 8);
                ub.u[1] = *(const uint4*)(vrow + k0 + 16 + hf * 8);
            } else {
                ub.u[0] = make_uint4(0, 0, 0, 0);
                ub.u[1] = make_uint4(0, 0, 0, 0);
            }
            c = __builtin_amdgcn_wmma_f32_16x16x32_f16(
                false, ua.v, false, ub.v, (short)0, c, false, false);
        }
        if (mn < 8) {
#pragma unroll
            for (int j = 0; j < 8; ++j) {
                int m = j + 8 * hf;
                int p = qb * 32 + wave * 16 + m;
                attn_pre[(size_t)(b * 64 + h * 8 + mn) * 1024 + p] = c[j];
            }
        }
    }
}

// ---------------------------------------------------------------------------
// Kernel 5: 1x1 output projection. GEMM M=8192, N=64, K=64 (f32 WMMA).
// Writes channels 64..127 of d_out (+bias).
// ---------------------------------------------------------------------------
__global__ void proj_kernel(const float* __restrict__ attn_pre,
                            const float* __restrict__ w_attn,
                            const float* __restrict__ b_attn,
                            float* __restrict__ out) {
    int tid  = blockIdx.x * blockDim.x + threadIdx.x;
    int wid  = tid >> 5;
    int lane = threadIdx.x & 31;
    int mt = wid >> 2, nt = wid & 3;
    int hf = lane >> 4, mn = lane & 15;
    int p = mt * 16 + mn, b = p >> 10, pp = p & 1023;
    int col = nt * 16 + mn;

    v8f c = {};
    for (int kt = 0; kt < 16; ++kt) {
        int k0 = kt * 4 + hf * 2;
        v2f a, bb;
        a.x  = attn_pre[(size_t)(b * 64 + k0) * 1024 + pp];
        a.y  = attn_pre[(size_t)(b * 64 + k0 + 1) * 1024 + pp];
        bb.x = w_attn[col * 64 + k0];
        bb.y = w_attn[col * 64 + k0 + 1];
        c = __builtin_amdgcn_wmma_f32_16x16x4_f32(
            false, a, false, bb, (short)0, c, false, false);
    }
#pragma unroll
    for (int j = 0; j < 8; ++j) {
        int m = j + 8 * hf;
        int P = mt * 16 + m, bb2 = P >> 10, pr = P & 1023;
        out[(size_t)(bb2 * 128 + 64 + col) * 1024 + pr] = c[j] + b_attn[col];
    }
}

// ---------------------------------------------------------------------------
extern "C" void kernel_launch(void* const* d_in, const int* in_sizes, int n_in,
                              void* d_out, int out_size, void* d_ws, size_t ws_size,
                              hipStream_t stream) {
    const float* x    = (const float*)d_in[0];
    const float* wco  = (const float*)d_in[1];
    const float* bco  = (const float*)d_in[2];
    const float* wqkv = (const float*)d_in[3];
    const float* bqkv = (const float*)d_in[4];
    const float* watt = (const float*)d_in[5];
    const float* batt = (const float*)d_in[6];
    const float* krw  = (const float*)d_in[7];
    const float* krh  = (const float*)d_in[8];
    float* out = (float*)d_out;

    char* ws = (char*)d_ws;
    size_t o = 0;
    _Float16* wfrag = (_Float16*)(ws + o); o += (size_t)147456 * 2;    // 288 KB
    o = (o + 255) & ~(size_t)255;
    _Float16* afrag = (_Float16*)(ws + o); o += (size_t)4718592 * 2;   // 9 MB
    o = (o + 255) & ~(size_t)255;
    float* q_ws     = (float*)(ws + o);    o += (size_t)524288 * 4;    // 2 MB
    float* k_ws     = (float*)(ws + o);    o += (size_t)524288 * 4;    // 2 MB
    _Float16* v_ws  = (_Float16*)(ws + o); o += (size_t)524288 * 2;    // 1 MB
    o = (o + 255) & ~(size_t)255;
    float* RW       = (float*)(ws + o);    o += (size_t)2097152 * 4;   // 8 MB
    float* RH       = (float*)(ws + o);    o += (size_t)2097152 * 4;   // 8 MB
    float* attn_pre = (float*)(ws + o);    o += (size_t)524288 * 4;    // 2 MB

    // 1) weight swizzle + im2col in fragment order
    prep_w_kernel<<<(147456 + 255) / 256, 256, 0, stream>>>(wco, wqkv, wfrag);
    prep_a_kernel<<<(4718592 + 255) / 256, 256, 0, stream>>>(x, afrag);

    // 2) fused 3x3 convs as implicit GEMM (f16 WMMA + async LDS staging)
    conv_gemm_kernel<<<256, 256, 0, stream>>>(afrag, wfrag, bco, bqkv,
                                              out, q_ws, k_ws, v_ws);

    // 3) relative logits RW / RH
    rel_kernel<<<16384, 256, 0, stream>>>(q_ws, krw, krh, RW, RH);

    // 4) attention (192 KB dynamic LDS per WG)
    (void)hipFuncSetAttribute((const void*)attn_kernel,
                              hipFuncAttributeMaxDynamicSharedMemorySize, 196608);
    attn_kernel<<<2048, 64, 196608, stream>>>(q_ws, k_ws, v_ws, RW, RH, attn_pre);

    // 5) 1x1 projection (f32 WMMA)
    proj_kernel<<<256, 256, 0, stream>>>(attn_pre, watt, batt, out);
}
